// _NeuralCE_Ising_Legacy_82712480186789
// MI455X (gfx1250) — matrix operand order repaired and checked by reading, not compiled, not run on hardware
//
#include <hip/hip_runtime.h>

typedef __attribute__((ext_vector_type(16))) _Float16 v16h;
typedef __attribute__((ext_vector_type(8)))  _Float16 v8h;
typedef __attribute__((ext_vector_type(8)))  float    v8f;

#define NF    128   // atom feature length F
#define NEDGE 41    // raw edge feature length
#define EPAD  64    // padded edge K
#define NBR   12    // neighbors per atom
#define AB    16    // atoms per block
#define ORIGF 92    // original atom feature dim
#define OPAD  96    // padded embed K
#define HF    256   // hidden readout dim

// ---------------------------------------------------------------------------
// CDNA5 async global->LDS copy (ASYNCcnt-tracked DMA, ISA 15.18.3 op 98).
// VDST VGPR carries the wave-relative LDS byte offset; generic shared-pointer
// low 32 bits are exactly that offset (flat->LDS truncates to addr[31:0]).
// ---------------------------------------------------------------------------
__device__ __forceinline__ void async_ld_b128(void* lds_dst, const void* gsrc) {
    unsigned off = (unsigned)(uintptr_t)lds_dst;
    asm volatile("global_load_async_to_lds_b128 %0, %1, off"
                 :: "v"(off), "v"(gsrc)
                 : "memory");
}
__device__ __forceinline__ void wait_async() {
    asm volatile("s_wait_asynccnt 0x0" ::: "memory");
}

// ---------------------------------------------------------------------------
// WMMA fragment loaders (CDNA5 16x16x32 f16 layouts, ISA 7.12.2)
// A (16x32, row-major source, ldk halves):
//   lane<16 : row=lane,    K = {0..7, 16..23}  (+k0)
//   lane>=16: row=lane-16, K = {8..15, 24..31} (+k0)
// B from [N][K] (transposed weights): lane<16: n=lane, K = {0..7, 16..23};
//   lane>=16: n=lane-16, K = {8..15, 24..31}.
// ---------------------------------------------------------------------------
__device__ __forceinline__ v16h frag_a(const _Float16* base, int row0, int ldk, int k0) {
    const int lane = threadIdx.x & 31;
    const int r  = row0 + (lane & 15);
    const int kb = k0 + ((lane < 16) ? 0 : 8);
    const _Float16* p = base + (size_t)r * ldk + kb;
    v8h lo = *(const v8h*)(p);
    v8h hi = *(const v8h*)(p + 16);
    v16h o;
#pragma unroll
    for (int i = 0; i < 8; ++i) { o[i] = lo[i]; o[i + 8] = hi[i]; }
    return o;
}

__device__ __forceinline__ v16h frag_b(const _Float16* base, int n0, int ldk, int k0) {
    const int lane = threadIdx.x & 31;
    const int n  = n0 + (lane & 15);
    const int kb = k0 + ((lane < 16) ? 0 : 8);
    const _Float16* p = base + (size_t)n * ldk + kb;
    v8h lo = *(const v8h*)(p);
    v8h hi = *(const v8h*)(p + 16);
    v16h o;
#pragma unroll
    for (int i = 0; i < 8; ++i) { o[i] = lo[i]; o[i + 8] = hi[i]; }
    return o;
}

__device__ __forceinline__ v8f wmma16(v16h a, v16h b, v8f c) {
    return __builtin_amdgcn_wmma_f32_16x16x32_f16(
        /*neg_a=*/false, a, /*neg_b=*/false, b,
        /*c_mod=*/(short)0, c, /*reuse_a=*/false, /*reuse_b=*/false);
}

__device__ __forceinline__ float dev_sigmoid(float x) { return 1.0f / (1.0f + __expf(-x)); }
__device__ __forceinline__ float dev_softplus(float x) {
    return (x > 20.0f) ? x : __logf(1.0f + __expf(x));
}

// ---------------------------------------------------------------------------
// Weight prep: out[n*Kpad + k] = (k<K) ? W[k*N + n] : 0   (fp32 -> f16, transposed)
// ---------------------------------------------------------------------------
__global__ void cvt_transpose_kernel(const float* __restrict__ W, _Float16* __restrict__ out,
                                     int K, int N, int Kpad) {
    int i = blockIdx.x * blockDim.x + threadIdx.x;
    int total = N * Kpad;
    if (i >= total) return;
    int n = i / Kpad;
    int k = i - n * Kpad;
    out[i] = (k < K) ? (_Float16)W[(size_t)k * N + n] : (_Float16)0.0f;
}

// ---------------------------------------------------------------------------
// Embedding: x = atom_fea @ W_embed + b  (K 92 padded to 96). 16 atoms/block.
// ---------------------------------------------------------------------------
__global__ __launch_bounds__(64) void embed_kernel(
    const float* __restrict__ atom_fea, const _Float16* __restrict__ Wemb_t,
    const float* __restrict__ b_embed, float* __restrict__ xf32,
    _Float16* __restrict__ xf16) {
    __shared__ _Float16 sA[AB * OPAD];
    const int tid  = threadIdx.x;
    const int wave = tid >> 5;
    const int lane = tid & 31;
    const int atom0 = blockIdx.x * AB;

    for (int i = tid; i < AB * OPAD; i += 64) {
        int r = i / OPAD, k = i - r * OPAD;
        sA[i] = (k < ORIGF) ? (_Float16)atom_fea[(size_t)(atom0 + r) * ORIGF + k]
                            : (_Float16)0.0f;
    }
    __syncthreads();

    for (int nt = wave; nt < 8; nt += 2) {
        v8f acc = {};
#pragma unroll
        for (int ks = 0; ks < 3; ++ks) {
            v16h a = frag_a(sA, 0, OPAD, ks * 32);
            v16h b = frag_b(Wemb_t, nt * 16, OPAD, ks * 32);
            acc = wmma16(a, b, acc);
        }
        const int col   = nt * 16 + (lane & 15);
        const float bia = b_embed[col];
        const int rbase = (lane < 16) ? 0 : 8;
#pragma unroll
        for (int r = 0; r < 8; ++r) {
            int row = r + rbase;
            float v = acc[r] + bia;
            xf32[(size_t)(atom0 + row) * NF + col] = v;
            xf16[(size_t)(atom0 + row) * NF + col] = (_Float16)v;
        }
    }
}

// ---------------------------------------------------------------------------
// Fused conv layer. 16 atoms/block, 192 threads (6 waves), neighbors in 4
// chunks of 3 (48 WMMA rows per chunk; row = nbl*16 + local_atom).
// ---------------------------------------------------------------------------
__global__ __launch_bounds__(192) void conv_kernel(
    const float* __restrict__ xf32_in, const _Float16* __restrict__ xf16_in,
    float* __restrict__ xf32_out, _Float16* __restrict__ xf16_out,
    const float* __restrict__ nbr_fea, const int* __restrict__ nbr_idx,
    const _Float16* __restrict__ Wc_t, const float* __restrict__ b_c,
    const _Float16* __restrict__ Wn_t, const float* __restrict__ b_n,
    const _Float16* __restrict__ We_t, const float* __restrict__ b_e,
    const _Float16* __restrict__ Wg_t, const float* __restrict__ b_g,
    const _Float16* __restrict__ Wm_t, const float* __restrict__ b_m,
    const float* __restrict__ ln_scale, const float* __restrict__ ln_bias) {

    __shared__ _Float16 sX[AB * NF];       //  4 KB  block atoms, f16
    __shared__ _Float16 sAn[48 * NF];      // 12 KB  gathered neighbor feats
    __shared__ _Float16 sAe[48 * EPAD];    //  6 KB  edge feats (padded)
    __shared__ _Float16 sInter[48 * NF];   // 12 KB  phi_e, then inter
    __shared__ float    sPhiC[AB * NF];    //  8 KB
    __shared__ float    sAcc[AB * NF];     //  8 KB  sum over neighbors

    const int tid  = threadIdx.x;
    const int wave = tid >> 5;             // 0..5
    const int lane = tid & 31;
    const int atom0 = blockIdx.x * AB;

    // stage block atom features via async DMA; zero accumulator
    for (int i = tid; i < (AB * NF) / 8; i += 192)
        async_ld_b128((uint4*)sX + i, (const uint4*)(xf16_in + (size_t)atom0 * NF) + i);
    for (int i = tid; i < AB * NF; i += 192) sAcc[i] = 0.0f;

    // hint the L2-resident weights toward the WGP
    if (tid == 0) {
        __builtin_prefetch(Wg_t, 0, 0);
        __builtin_prefetch(Wm_t, 0, 0);
        __builtin_prefetch(Wn_t, 0, 0);
    }
    wait_async();
    __syncthreads();

    // phi_c = x @ W_center + b_center  -> sPhiC (fp32)
    for (int nt = wave; nt < 8; nt += 6) {
        v8f acc = {};
#pragma unroll
        for (int ks = 0; ks < 4; ++ks) {
            v16h a = frag_a(sX, 0, NF, ks * 32);
            v16h b = frag_b(Wc_t, nt * 16, NF, ks * 32);
            acc = wmma16(a, b, acc);
        }
        const int col = nt * 16 + (lane & 15);
        const float bia = b_c[col];
        const int rbase = (lane < 16) ? 0 : 8;
#pragma unroll
        for (int r = 0; r < 8; ++r) sPhiC[(r + rbase) * NF + col] = acc[r] + bia;
    }

    const int mt  = wave >> 1;        // M-tile (neighbor-in-chunk) 0..2
    const int nt0 = (wave & 1) * 4;   // N-tile start (0 or 4)

    for (int chunk = 0; chunk < 4; ++chunk) {
        __syncthreads();  // prev chunk fully consumed; phi_c ready on chunk 0

        // gather neighbor atom features: 48 rows x 256B, async DMA to LDS
        for (int i = tid; i < 48 * (NF / 8); i += 192) {
            int lr  = i >> 4;          // row 0..47
            int seg = i & 15;          // uint4 segment
            int a   = lr & 15;
            int nbl = lr >> 4;
            int m   = chunk * 3 + nbl;
            int gidx = nbr_idx[(size_t)(atom0 + a) * NBR + m];
            async_ld_b128((uint4*)sAn + i,
                          (const uint4*)(xf16_in + (size_t)gidx * NF) + seg);
        }
        // edge features fp32 -> f16 (needs convert, stays on the VALU path)
        for (int i = tid; i < 48 * EPAD; i += 192) {
            int lr = i >> 6;
            int k  = i & 63;
            int a   = lr & 15;
            int nbl = lr >> 4;
            int m   = chunk * 3 + nbl;
            float v = (k < NEDGE)
                ? nbr_fea[((size_t)(atom0 + a) * NBR + m) * NEDGE + k] : 0.0f;
            sAe[i] = (_Float16)v;
        }
        wait_async();
        __syncthreads();

        // phi_e = edge @ W_edge + b_edge  -> sInter (f16), K = 64
        for (int ntl = 0; ntl < 4; ++ntl) {
            const int nt = nt0 + ntl;
            v8f acc = {};
#pragma unroll
            for (int ks = 0; ks < 2; ++ks) {
                v16h a = frag_a(sAe, mt * 16, EPAD, ks * 32);
                v16h b = frag_b(We_t, nt * 16, EPAD, ks * 32);
                acc = wmma16(a, b, acc);
            }
            const int col = nt * 16 + (lane & 15);
            const float be = b_e[col];
            const int rbase = (lane < 16) ? 0 : 8;
#pragma unroll
            for (int r = 0; r < 8; ++r)
                sInter[(mt * 16 + r + rbase) * NF + col] = (_Float16)(acc[r] + be);
        }
        // no barrier needed: each wave reads back only tiles it wrote

        // phi_n = gathered @ W_nbr + b_nbr; inter = phi_c * phi_n * phi_e
        for (int ntl = 0; ntl < 4; ++ntl) {
            const int nt = nt0 + ntl;
            v8f acc = {};
#pragma unroll
            for (int ks = 0; ks < 4; ++ks) {
                v16h a = frag_a(sAn, mt * 16, NF, ks * 32);
                v16h b = frag_b(Wn_t, nt * 16, NF, ks * 32);
                acc = wmma16(a, b, acc);
            }
            const int col = nt * 16 + (lane & 15);
            const float bn = b_n[col];
            const int rbase = (lane < 16) ? 0 : 8;
#pragma unroll
            for (int r = 0; r < 8; ++r) {
                int row   = mt * 16 + r + rbase;
                int aidx  = r + rbase;              // row & 15
                float pn  = acc[r] + bn;
                float pe  = (float)sInter[row * NF + col];
                float pc  = sPhiC[aidx * NF + col];
                sInter[row * NF + col] = (_Float16)(pc * pn * pe);
            }
        }
        __syncthreads();  // full inter rows needed as K for gate/mag

        // gate/mag GEMMs over inter (K=128), activation, neighbor-sum
        for (int ntl = 0; ntl < 4; ++ntl) {
            const int nt = nt0 + ntl;
            v8f accg = {}, accm = {};
#pragma unroll
            for (int ks = 0; ks < 4; ++ks) {
                v16h a  = frag_a(sInter, mt * 16, NF, ks * 32);
                v16h bg = frag_b(Wg_t, nt * 16, NF, ks * 32);
                accg = wmma16(a, bg, accg);
                v16h bm = frag_b(Wm_t, nt * 16, NF, ks * 32);
                accm = wmma16(a, bm, accm);
            }
            const int col = nt * 16 + (lane & 15);
            const float bg = b_g[col], bm = b_m[col];
            const int rbase = (lane < 16) ? 0 : 8;
#pragma unroll
            for (int r = 0; r < 8; ++r) {
                int aidx = r + rbase;
                float g  = dev_sigmoid(accg[r] + bg);
                float sp = dev_softplus(accm[r] + bm);
                atomicAdd(&sAcc[aidx * NF + col], g * sp);
            }
        }
    }
    __syncthreads();

    // LayerNorm + residual (one thread per atom)
    if (tid < AB) {
        const int a = tid;
        float mu = 0.0f;
        for (int f = 0; f < NF; ++f) mu += sAcc[a * NF + f];
        mu *= (1.0f / NF);
        float var = 0.0f;
        for (int f = 0; f < NF; ++f) {
            float d = sAcc[a * NF + f] - mu;
            var += d * d;
        }
        var *= (1.0f / NF);
        const float rstd = rsqrtf(var + 1e-6f);
        for (int f = 0; f < NF; ++f) {
            float ln = (sAcc[a * NF + f] - mu) * rstd * ln_scale[f] + ln_bias[f];
            float nx = xf32_in[(size_t)(atom0 + a) * NF + f] + ln;
            xf32_out[(size_t)(atom0 + a) * NF + f] = nx;
            xf16_out[(size_t)(atom0 + a) * NF + f] = (_Float16)nx;
        }
    }
}

// ---------------------------------------------------------------------------
// Readout: per-block (16 atoms) partial of sum softplus(x@W_h + b_h) @ W_out.
// ---------------------------------------------------------------------------
__global__ __launch_bounds__(128) void readout_kernel(
    const _Float16* __restrict__ xf16, const _Float16* __restrict__ Wh_t,
    const float* __restrict__ b_h, const float* __restrict__ W_out,
    const float* __restrict__ b_out, float* __restrict__ partials) {
    __shared__ _Float16 sA[AB * NF];
    __shared__ float sE[AB];
    const int tid  = threadIdx.x;
    const int wave = tid >> 5;
    const int lane = tid & 31;
    const int atom0 = blockIdx.x * AB;

    for (int i = tid; i < (AB * NF) / 8; i += 128)
        async_ld_b128((uint4*)sA + i, (const uint4*)(xf16 + (size_t)atom0 * NF) + i);
    if (tid < AB) sE[tid] = 0.0f;
    wait_async();
    __syncthreads();

    for (int nt = wave; nt < HF / 16; nt += 4) {
        v8f acc = {};
#pragma unroll
        for (int ks = 0; ks < 4; ++ks) {
            v16h a = frag_a(sA, 0, NF, ks * 32);
            v16h b = frag_b(Wh_t, nt * 16, NF, ks * 32);
            acc = wmma16(a, b, acc);
        }
        const int col = nt * 16 + (lane & 15);
        const float bh = b_h[col];
        const float wo = W_out[col];
        const int rbase = (lane < 16) ? 0 : 8;
#pragma unroll
        for (int r = 0; r < 8; ++r)
            atomicAdd(&sE[r + rbase], dev_softplus(acc[r] + bh) * wo);
    }
    __syncthreads();
    if (tid == 0) {
        float s = 0.0f;
        for (int a = 0; a < AB; ++a) s += sE[a];
        partials[blockIdx.x] = s + (float)AB * b_out[0];
    }
}

__global__ void reduce_kernel(const float* __restrict__ partials, float* __restrict__ out,
                              int blocks_per_crystal, int n_crystal) {
    int c = blockIdx.x * blockDim.x + threadIdx.x;
    if (c >= n_crystal) return;
    float s = 0.0f;
    for (int i = 0; i < blocks_per_crystal; ++i)
        s += partials[(size_t)c * blocks_per_crystal + i];
    out[c] = s;
}

// ---------------------------------------------------------------------------
extern "C" void kernel_launch(void* const* d_in, const int* in_sizes, int n_in,
                              void* d_out, int out_size, void* d_ws, size_t ws_size,
                              hipStream_t stream) {
    const float* atom_fea = (const float*)d_in[0];
    const float* nbr_fea  = (const float*)d_in[1];
    const int*   nbr_idx  = (const int*)d_in[2];
    const float* W_embed  = (const float*)d_in[3];
    const float* b_embed  = (const float*)d_in[4];
    const float* W_center = (const float*)d_in[5];
    const float* b_center = (const float*)d_in[6];
    const float* W_nbr    = (const float*)d_in[7];
    const float* b_nbr    = (const float*)d_in[8];
    const float* W_edge   = (const float*)d_in[9];
    const float* b_edge   = (const float*)d_in[10];
    const float* W_gate   = (const float*)d_in[11];
    const float* b_gate   = (const float*)d_in[12];
    const float* W_mag    = (const float*)d_in[13];
    const float* b_mag    = (const float*)d_in[14];
    const float* ln_scale = (const float*)d_in[15];
    const float* ln_bias  = (const float*)d_in[16];
    const float* W_h      = (const float*)d_in[17];
    const float* b_h      = (const float*)d_in[18];
    const float* W_out    = (const float*)d_in[19];
    const float* b_out    = (const float*)d_in[20];

    const int n_atoms = in_sizes[0] / ORIGF;      // 32768
    const int nblk    = n_atoms / AB;             // 2048
    const int n_cry   = out_size;                 // 128
    const int bpc     = nblk / (n_cry > 0 ? n_cry : 1);

    char* ws = (char*)d_ws;
    size_t off = 0;
    auto alloc = [&](size_t bytes) -> void* {
        off = (off + 255) & ~(size_t)255;
        void* p = ws + off;
        off += bytes;
        return p;
    };
    float*    x32[2];
    _Float16* x16[2];
    x32[0] = (float*)alloc((size_t)n_atoms * NF * 4);
    x32[1] = (float*)alloc((size_t)n_atoms * NF * 4);
    x16[0] = (_Float16*)alloc((size_t)n_atoms * NF * 2);
    x16[1] = (_Float16*)alloc((size_t)n_atoms * NF * 2);
    _Float16* wemb_t = (_Float16*)alloc((size_t)NF * OPAD * 2);
    _Float16* wc_t   = (_Float16*)alloc((size_t)3 * NF * NF * 2);
    _Float16* wn_t   = (_Float16*)alloc((size_t)3 * NF * NF * 2);
    _Float16* wg_t   = (_Float16*)alloc((size_t)3 * NF * NF * 2);
    _Float16* wm_t   = (_Float16*)alloc((size_t)3 * NF * NF * 2);
    _Float16* we_t   = (_Float16*)alloc((size_t)3 * NF * EPAD * 2);
    _Float16* wh_t   = (_Float16*)alloc((size_t)HF * NF * 2);
    float* partials  = (float*)alloc((size_t)nblk * 4);

    auto cvt = [&](const float* W, _Float16* o, int K, int N, int Kpad) {
        int tot = N * Kpad;
        cvt_transpose_kernel<<<(tot + 255) / 256, 256, 0, stream>>>(W, o, K, N, Kpad);
    };
    cvt(W_embed, wemb_t, ORIGF, NF, OPAD);
    for (int l = 0; l < 3; ++l) {
        cvt(W_center + (size_t)l * NF * NF, wc_t + (size_t)l * NF * NF, NF, NF, NF);
        cvt(W_nbr    + (size_t)l * NF * NF, wn_t + (size_t)l * NF * NF, NF, NF, NF);
        cvt(W_gate   + (size_t)l * NF * NF, wg_t + (size_t)l * NF * NF, NF, NF, NF);
        cvt(W_mag    + (size_t)l * NF * NF, wm_t + (size_t)l * NF * NF, NF, NF, NF);
        cvt(W_edge   + (size_t)l * NEDGE * NF, we_t + (size_t)l * NF * EPAD, NEDGE, NF, EPAD);
    }
    cvt(W_h, wh_t, NF, HF, NF);

    embed_kernel<<<nblk, 64, 0, stream>>>(atom_fea, wemb_t, b_embed, x32[0], x16[0]);

    int cur = 0;
    for (int l = 0; l < 3; ++l) {
        conv_kernel<<<nblk, 192, 0, stream>>>(
            x32[cur], x16[cur], x32[1 - cur], x16[1 - cur], nbr_fea, nbr_idx,
            wc_t + (size_t)l * NF * NF,  b_center + (size_t)l * NF,
            wn_t + (size_t)l * NF * NF,  b_nbr    + (size_t)l * NF,
            we_t + (size_t)l * NF * EPAD, b_edge  + (size_t)l * NF,
            wg_t + (size_t)l * NF * NF,  b_gate   + (size_t)l * NF,
            wm_t + (size_t)l * NF * NF,  b_mag    + (size_t)l * NF,
            ln_scale + (size_t)l * NF,   ln_bias  + (size_t)l * NF);
        cur = 1 - cur;
    }

    readout_kernel<<<nblk, 128, 0, stream>>>(x16[cur], wh_t, b_h, W_out, b_out, partials);
    reduce_kernel<<<(n_cry + 127) / 128, 128, 0, stream>>>(partials, (float*)d_out, bpc, n_cry);
    (void)n_in; (void)ws_size;
}
